// Mamba2DBlock_69097433858256
// MI455X (gfx1250) — matrix operand order repaired and checked by reading, not compiled, not run on hardware
//
#include <hip/hip_runtime.h>
#include <hip/hip_bf16.h>

typedef __attribute__((ext_vector_type(16))) __bf16 v16bf;
typedef __attribute__((ext_vector_type(8)))  float  v8f;
typedef __attribute__((ext_vector_type(4)))  int    v4i;

#define M_TOK 8192   // 128 sequences * 64 tokens per pass

#if defined(__HIP_DEVICE_COMPILE__) && __has_builtin(__builtin_amdgcn_global_load_async_to_lds_b128)
#define HAS_ASYNC_CP 1
#else
#define HAS_ASYNC_CP 0
#endif

#if HAS_ASYNC_CP
#define ASG __attribute__((address_space(1)))
#define ASL __attribute__((address_space(3)))
__device__ __forceinline__ void async_cp16(const void* g, void* l) {
  __builtin_amdgcn_global_load_async_to_lds_b128(
      (ASG v4i*)const_cast<void*>(g), (ASL v4i*)l, 0, 0);
}
#endif

template <int N>
__device__ __forceinline__ void wait_async() {
#if defined(__HIP_DEVICE_COMPILE__)
#if __has_builtin(__builtin_amdgcn_s_wait_asynccnt)
  __builtin_amdgcn_s_wait_asynccnt(N);
#else
  asm volatile("s_wait_asynccnt %0" ::"n"(N) : "memory");
#endif
#endif
}

// rate-10 strided reorder for L=64: idx[p'] = original position
__device__ __constant__ int c_idx[64] = {
  0,10,20,30,40,50,60, 1,11,21,31,41,51,61, 2,12,22,32,42,52,62,
  3,13,23,33,43,53,63, 4,14,24,34,44,54, 5,15,25,35,45,55,
  6,16,26,36,46,56, 7,17,27,37,47,57, 8,18,28,38,48,58, 9,19,29,39,49,59 };

__device__ __forceinline__ int src_token(int r, int dir) {
  int seq = r >> 6, pp = r & 63, pos = c_idx[pp];
  // dir 0: seq = b*64+h, token = (b*64+h)*64 + pos   (row scan over W)
  // dir 1: seq = b*64+w, token = ((b*64+pos)*64) + w (col scan over H)
  return (dir == 0) ? (seq * 64 + pos)
                    : (((seq >> 6) * 64 + pos) * 64 + (seq & 63));
}

// ---------------- fp32 -> bf16 convert (weights / small tensors) ----------
__global__ void k_f32_to_bf16(const float* __restrict__ s,
                              unsigned short* __restrict__ d, int n) {
  int i = blockIdx.x * 256 + threadIdx.x;
  if (i < n) ((__bf16*)d)[i] = (__bf16)s[i];
}

// ---------------- gather tokens (apply rate-10 permutation) + convert -----
__global__ void k_gather_bf16(const float* __restrict__ src,
                              unsigned short* __restrict__ dst, int dir) {
  int i = blockIdx.x * 256 + threadIdx.x;          // M_TOK*512 threads
  int r = i >> 9, d = i & 511;
  int tok = src_token(r, dir);
  ((__bf16*)dst)[(size_t)r * 512 + d] = (__bf16)src[(size_t)tok * 512 + d];
}

// ---------------- WMMA bf16 GEMM: C(MxN) = A(MxK) * W(NxK)^T --------------
// workgroup tile 128x64, K-step 32, double-buffered LDS, async global->LDS.
// 8 waves: wave w -> rows [w*16, w*16+16), 4 N-tiles each (4 x v8f acc).
// mode 0: C = acc ; mode 1: C = softplus(acc + bias[n]) ; mode 2: scattered
__global__ __launch_bounds__(256) void k_gemm_bf16(
    const unsigned short* __restrict__ Ain, const unsigned short* __restrict__ Bin,
    float* __restrict__ C, int M, int N, int K,
    int mode, const float* __restrict__ bias, int dir) {
  constexpr int LDA = 40, LDB = 40;                // bf16 elems, pad vs banks
  __shared__ __align__(16) __bf16 ldsA[2][128 * LDA];
  __shared__ __align__(16) __bf16 ldsB[2][64 * LDB];
  const __bf16* A = (const __bf16*)Ain;
  const __bf16* B = (const __bf16*)Bin;

  const int m0 = blockIdx.x * 128;
  const int n0 = blockIdx.y * 64;
  const int t = threadIdx.x;
  const int lane = t & 31, wv = t >> 5;
  const int half = lane >> 4, l16 = lane & 15;

  v8f acc[4] = {v8f{}, v8f{}, v8f{}, v8f{}};

  const int arow = t >> 1, ah = (t & 1) * 16;      // A stage: 16 elems/thread
  const int brow = t >> 2, bq = (t & 3) * 8;       // B stage: 8 elems/thread
  const __bf16* gA = A + (size_t)(m0 + arow) * K + ah;
  const __bf16* gB = B + (size_t)(n0 + brow) * K + bq;

#if HAS_ASYNC_CP
  // prologue: fill buffer 0 for k0 = 0 (3 async instructions per wave)
  async_cp16(gA, &ldsA[0][arow * LDA + ah]);
  async_cp16(gA + 8, &ldsA[0][arow * LDA + ah + 8]);
  async_cp16(gB, &ldsB[0][brow * LDB + bq]);
#endif

  int buf = 0;
  for (int k0 = 0; k0 < K; k0 += 32, buf ^= 1) {
#if HAS_ASYNC_CP
    const bool more = (k0 + 32) < K;
    if (more) {                                    // prefetch next tile
      async_cp16(gA + k0 + 32, &ldsA[buf ^ 1][arow * LDA + ah]);
      async_cp16(gA + k0 + 40, &ldsA[buf ^ 1][arow * LDA + ah + 8]);
      async_cp16(gB + k0 + 32, &ldsB[buf ^ 1][brow * LDB + bq]);
      wait_async<3>();                             // current batch complete
    } else {
      wait_async<0>();
    }
    __syncthreads();                               // all waves' fills visible
#else
    uint4 a0 = *reinterpret_cast<const uint4*>(gA + k0);
    uint4 a1 = *reinterpret_cast<const uint4*>(gA + k0 + 8);
    uint4 b0 = *reinterpret_cast<const uint4*>(gB + k0);
    __syncthreads();
    *reinterpret_cast<uint4*>(&ldsA[buf][arow * LDA + ah])     = a0;
    *reinterpret_cast<uint4*>(&ldsA[buf][arow * LDA + ah + 8]) = a1;
    *reinterpret_cast<uint4*>(&ldsB[buf][brow * LDB + bq])     = b0;
    __syncthreads();
#endif

    // A fragment (16x32, ISA layout): lane half picks K base; two b128 reads
    union { v16bf v; uint4 u[2]; } af;
    const __bf16* pa = &ldsA[buf][(wv * 16 + l16) * LDA + 8 * half];
    af.u[0] = *reinterpret_cast<const uint4*>(pa);        // K = 8h .. 8h+7
    af.u[1] = *reinterpret_cast<const uint4*>(pa + 16);   // K = 16+8h ..

#pragma unroll
    for (int nt = 0; nt < 4; ++nt) {
      union { v16bf v; uint4 u[2]; } bfr;
      const __bf16* pb = &ldsB[buf][(nt * 16 + l16) * LDB + 8 * half];
      bfr.u[0] = *reinterpret_cast<const uint4*>(pb);     // K = 8h .. 8h+7
      bfr.u[1] = *reinterpret_cast<const uint4*>(pb + 16); // K = 16+8h ..
      acc[nt] = __builtin_amdgcn_wmma_f32_16x16x32_bf16(
          false, af.v, false, bfr.v, (short)0, acc[nt], false, false);
    }
    __syncthreads();                               // reads done before refill
  }

#pragma unroll
  for (int nt = 0; nt < 4; ++nt) {
#pragma unroll
    for (int r = 0; r < 8; ++r) {
      int gr = m0 + wv * 16 + r + 8 * half;   // C layout: VGPR r -> M=r / 8+r
      int gc = n0 + nt * 16 + l16;
      float v = acc[nt][r];
      if (mode == 0) {
        C[(size_t)gr * N + gc] = v;
      } else if (mode == 1) {
        float s = v + bias[gc];
        C[(size_t)gr * N + gc] = (s > 20.f) ? s : log1pf(__expf(s));
      } else {
        int tok = src_token(gr, dir);         // undo reorder (scatter)
        C[(size_t)tok * N + gc] = v;
      }
    }
  }
}

// ---------------- depthwise causal conv(4) + SiLU -------------------------
__global__ void k_conv_silu(const float* __restrict__ xz,
                            const float* __restrict__ cw, const float* __restrict__ cb,
                            float* __restrict__ xc, unsigned short* __restrict__ xcbf) {
  int i = blockIdx.x * 256 + threadIdx.x;          // M_TOK*1024
  int r = i >> 10, e = i & 1023;
  int p = r & 63, rb = r - p;
  float acc = cb[e];
#pragma unroll
  for (int j = 0; j < 4; ++j) {
    int q = p - 3 + j;
    if (q >= 0) acc += cw[e * 4 + j] * xz[(size_t)(rb + q) * 2048 + e];
  }
  float s = acc / (1.f + __expf(-acc));
  xc[(size_t)r * 1024 + e] = s;
  ((__bf16*)xcbf)[(size_t)r * 1024 + e] = (__bf16)s;
}

// ---------------- extract dt low-rank part -> bf16 ------------------------
__global__ void k_dtlow(const float* __restrict__ xdbl, unsigned short* __restrict__ dtl) {
  int i = blockIdx.x * 256 + threadIdx.x;          // M_TOK*32
  int r = i >> 5, j = i & 31;
  ((__bf16*)dtl)[(size_t)r * 32 + j] = (__bf16)xdbl[(size_t)r * 64 + j];
}

// ---------------- selective scan: thread = (seq, channel), state in regs --
__global__ __launch_bounds__(256) void k_scan(
    const float* __restrict__ dt, const float* __restrict__ xdbl,
    const float* __restrict__ xc, const float* __restrict__ A_log,
    const float* __restrict__ Dp, float* __restrict__ ys) {
  __shared__ float sBC[32];                        // B_t[16] | C_t[16]
  int seq = blockIdx.x >> 2;
  int e = (blockIdx.x & 3) * 256 + threadIdx.x;
  float a[16], h[16];
#pragma unroll
  for (int n = 0; n < 16; ++n) { a[n] = -__expf(A_log[e * 16 + n]); h[n] = 0.f; }
  float dfac = Dp[e];
  for (int p = 0; p < 64; ++p) {
    int row = seq * 64 + p;
    __syncthreads();
    if (threadIdx.x < 32) sBC[threadIdx.x] = xdbl[(size_t)row * 64 + 32 + threadIdx.x];
    __syncthreads();
    float dtv = dt[(size_t)row * 1024 + e];
    float u   = xc[(size_t)row * 1024 + e];
    float du  = dtv * u;
    float y = 0.f;
#pragma unroll
    for (int n = 0; n < 16; ++n) {
      h[n] = __expf(dtv * a[n]) * h[n] + du * sBC[n];
      y += h[n] * sBC[16 + n];
    }
    ys[(size_t)row * 1024 + e] = y + dfac * u;
  }
}

// ---------------- gate: y = ys * silu(z), emit bf16 for out_proj ----------
__global__ void k_gate(const float* __restrict__ ys, const float* __restrict__ xz,
                       unsigned short* __restrict__ yg) {
  int i = blockIdx.x * 256 + threadIdx.x;
  int r = i >> 10, e = i & 1023;
  float z = xz[(size_t)r * 2048 + 1024 + e];
  float g = z / (1.f + __expf(-z));
  ((__bf16*)yg)[(size_t)r * 1024 + e] = (__bf16)(ys[(size_t)r * 1024 + e] * g);
}

// ---------------- residual + LayerNorm over D=512 -------------------------
__global__ __launch_bounds__(256) void k_layernorm(
    const float* __restrict__ yc, const float* __restrict__ x,
    const float* __restrict__ gamma, const float* __restrict__ beta,
    float* __restrict__ out) {
  __shared__ float s1[256], s2[256];
  int tok = blockIdx.x, t = threadIdx.x;
  float v0 = yc[(size_t)tok * 512 + t]       + x[(size_t)tok * 512 + t];
  float v1 = yc[(size_t)tok * 512 + 256 + t] + x[(size_t)tok * 512 + 256 + t];
  s1[t] = v0 + v1; s2[t] = v0 * v0 + v1 * v1;
  __syncthreads();
  for (int s = 128; s > 0; s >>= 1) {
    if (t < s) { s1[t] += s1[t + s]; s2[t] += s2[t + s]; }
    __syncthreads();
  }
  float mu = s1[0] * (1.f / 512.f);
  float var = s2[0] * (1.f / 512.f) - mu * mu;
  float rs = rsqrtf(var + 1e-5f);
  out[(size_t)tok * 512 + t]       = (v0 - mu) * rs * gamma[t]       + beta[t];
  out[(size_t)tok * 512 + 256 + t] = (v1 - mu) * rs * gamma[t + 256] + beta[t + 256];
}

extern "C" void kernel_launch(void* const* d_in, const int* in_sizes, int n_in,
                              void* d_out, int out_size, void* d_ws, size_t ws_size,
                              hipStream_t stream) {
  (void)in_sizes; (void)n_in; (void)out_size; (void)ws_size;
  const float* x     = (const float*)d_in[0];
  const float* gamma = (const float*)d_in[19];
  const float* beta  = (const float*)d_in[20];

  char* ws = (char*)d_ws;
  size_t o = 0;
  auto alloc = [&](size_t b) { size_t r = o; o += (b + 255) & ~(size_t)255; return r; };
  unsigned short* wWin  = (unsigned short*)(ws + alloc((size_t)2048 * 512 * 2));
  unsigned short* wWx   = (unsigned short*)(ws + alloc((size_t)64 * 1024 * 2));
  unsigned short* wWdt  = (unsigned short*)(ws + alloc((size_t)1024 * 32 * 2));
  unsigned short* wWout = (unsigned short*)(ws + alloc((size_t)512 * 1024 * 2));
  unsigned short* A0    = (unsigned short*)(ws + alloc((size_t)M_TOK * 512 * 2));
  float*          XZ    = (float*)(ws + alloc((size_t)M_TOK * 2048 * 4));
  float*          XC    = (float*)(ws + alloc((size_t)M_TOK * 1024 * 4));
  unsigned short* XCbf  = (unsigned short*)(ws + alloc((size_t)M_TOK * 1024 * 2));
  float*          XDBL  = (float*)(ws + alloc((size_t)M_TOK * 64 * 4));
  unsigned short* DTL   = (unsigned short*)(ws + alloc((size_t)M_TOK * 32 * 2));
  float*          DTm   = (float*)(ws + alloc((size_t)M_TOK * 1024 * 4));
  float*          YS    = (float*)(ws + alloc((size_t)M_TOK * 1024 * 4));
  unsigned short* YG    = (unsigned short*)(ws + alloc((size_t)M_TOK * 1024 * 2));
  float*          YROW  = (float*)(ws + alloc((size_t)M_TOK * 512 * 4));
  float*          YCOL  = (float*)(ws + alloc((size_t)M_TOK * 512 * 4));

  auto cvt = [&](const float* s, unsigned short* d, int n) {
    k_f32_to_bf16<<<(n + 255) / 256, 256, 0, stream>>>(s, d, n);
  };

  auto run_pass = [&](const float* src, float* dst, int dir, int wb) {
    const float* W_in  = (const float*)d_in[wb + 0];
    const float* convw = (const float*)d_in[wb + 1];
    const float* convb = (const float*)d_in[wb + 2];
    const float* W_x   = (const float*)d_in[wb + 3];
    const float* W_dt  = (const float*)d_in[wb + 4];
    const float* dtb   = (const float*)d_in[wb + 5];
    const float* A_log = (const float*)d_in[wb + 6];
    const float* Dp    = (const float*)d_in[wb + 7];
    const float* W_out = (const float*)d_in[wb + 8];

    cvt(W_in, wWin, 2048 * 512);
    cvt(W_x, wWx, 64 * 1024);
    cvt(W_dt, wWdt, 1024 * 32);
    cvt(W_out, wWout, 512 * 1024);
    k_gather_bf16<<<(M_TOK * 512) / 256, 256, 0, stream>>>(src, A0, dir);
    // in_proj: (8192x512)*(2048x512)^T
    k_gemm_bf16<<<dim3(64, 32), 256, 0, stream>>>(A0, wWin, XZ, M_TOK, 2048, 512, 0, nullptr, 0);
    k_conv_silu<<<(M_TOK * 1024) / 256, 256, 0, stream>>>(XZ, convw, convb, XC, XCbf);
    // x_proj: (8192x1024)*(64x1024)^T
    k_gemm_bf16<<<dim3(64, 1), 256, 0, stream>>>(XCbf, wWx, XDBL, M_TOK, 64, 1024, 0, nullptr, 0);
    k_dtlow<<<(M_TOK * 32) / 256, 256, 0, stream>>>(XDBL, DTL);
    // dt_proj + softplus: (8192x32)*(1024x32)^T
    k_gemm_bf16<<<dim3(64, 16), 256, 0, stream>>>(DTL, wWdt, DTm, M_TOK, 1024, 32, 1, dtb, 0);
    k_scan<<<512, 256, 0, stream>>>(DTm, XDBL, XC, A_log, Dp, YS);
    k_gate<<<(M_TOK * 1024) / 256, 256, 0, stream>>>(YS, XZ, YG);
    // out_proj + inverse-permute scatter: (8192x1024)*(512x1024)^T
    k_gemm_bf16<<<dim3(64, 8), 256, 0, stream>>>(YG, wWout, dst, M_TOK, 512, 1024, 2, nullptr, dir);
  };

  run_pass(x, YROW, 0, 1);       // row-direction SRMamba
  run_pass(YROW, YCOL, 1, 10);   // column-direction SRMamba
  k_layernorm<<<M_TOK, 256, 0, stream>>>(YCOL, x, gamma, beta, (float*)d_out);
}
